// GeometricPoseEstimator_28192165331077
// MI455X (gfx1250) — matrix operand order repaired and checked by reading, not compiled, not run on hardware
//
#include <hip/hip_runtime.h>
#include <hip/hip_bf16.h>
#include <math.h>

typedef __attribute__((ext_vector_type(16))) _Float16 v16h;
typedef __attribute__((ext_vector_type(8)))  float    v8f;

#define DEV __device__ __forceinline__

DEV v8f wmma16x16x32(v16h a, v16h b, v8f c) {
  return __builtin_amdgcn_wmma_f32_16x16x32_f16(false, a, false, b, (short)0, c, false, false);
}

// A fragment (M=16, K=32) from LDS tile stored row-major [rows][ld] halves.
// lane holds row = row_base + lane%16; element e -> k = e + (lane>=16?16:0)
DEV v16h load_a_frag(const _Float16* lds, int row_base, int koff, int ld, int lane) {
  int r = row_base + (lane & 15);
  int khi = (lane & 16) ? 16 : 0;
  const _Float16* p = lds + (size_t)r * ld + koff + khi;
  v16h a;
#pragma unroll
  for (int v = 0; v < 8; ++v) {
    int k = 2 * v;
    a[2 * v]     = p[k];
    a[2 * v + 1] = p[k + 1];
  }
  return a;
}

// B fragment (K=32, N=16) where LDS tile stores Bt[n][k] (weight row-major / K-tile [key][hd]).
// lane holds col n = col_base + lane%16; element e -> k = koff + e + 16*(lane>=16)
DEV v16h load_b_frag_nk(const _Float16* lds, int col_base, int koff, int ld, int lane) {
  int n = col_base + (lane & 15);
  int k0 = koff + ((lane & 16) ? 16 : 0);
  const _Float16* p = lds + (size_t)n * ld + k0;
  v16h b;
#pragma unroll
  for (int e = 0; e < 16; ++e) b[e] = p[e];
  return b;
}

// B fragment where LDS tile stores B[k][n] (V tile: [key][hd])
DEV v16h load_b_frag_kn(const _Float16* lds, int col_base, int koff, int ld, int lane) {
  int n = col_base + (lane & 15);
  int k0 = koff + ((lane & 16) ? 16 : 0);
  v16h b;
#pragma unroll
  for (int e = 0; e < 16; ++e) b[e] = lds[(size_t)(k0 + e) * ld + n];
  return b;
}

DEV float gelu_exact(float x) { return 0.5f * x * (1.f + erff(x * 0.70710678118654752f)); }

DEV float block_reduce_sum(float v, float* red, int nthreads) {
  int t = threadIdx.x;
  red[t] = v;
  __syncthreads();
  for (int off = nthreads >> 1; off > 0; off >>= 1) {
    if (t < off) red[t] += red[t + off];
    __syncthreads();
  }
  float r = red[0];
  __syncthreads();
  return r;
}

// ---------------------------------------------------------------------------
// Kernel 0: f32 -> f16 weight conversion (n multiple of 4)
// ---------------------------------------------------------------------------
__global__ void cvt_f16_kernel(const float* __restrict__ s, _Float16* __restrict__ d, int n) {
  int i = (blockIdx.x * 256 + threadIdx.x) * 4;
  if (i < n) {
    float4 v = *(const float4*)(s + i);
    union { _Float16 h[4]; uint2 u; } r;
    r.h[0] = (_Float16)v.x; r.h[1] = (_Float16)v.y;
    r.h[2] = (_Float16)v.z; r.h[3] = (_Float16)v.w;
    *(uint2*)(d + i) = r.u;
  }
}

// ---------------------------------------------------------------------------
// Kernel 1: per-batch selection of first 512 masked flat indices (ascending)
// mask = score > 0.3 * rowmax.  grid=B, block=1024
// ---------------------------------------------------------------------------
__global__ void topk_kernel(const float* __restrict__ scores,
                            int* __restrict__ i1o, int* __restrict__ i2o) {
  __shared__ float red[1024];
  __shared__ unsigned pref[1024];
  __shared__ int s_count;
  int b = blockIdx.x, t = threadIdx.x;
  if (t == 0) s_count = 0;
  __syncthreads();
  const float* srow = scores + (size_t)b * 1024 * 1024;
  for (int n1 = 0; n1 < 1024; ++n1) {
    int count = s_count;
    if (count >= 512) break;
    float s = srow[(size_t)n1 * 1024 + t];
    red[t] = s;
    __syncthreads();
    for (int off = 512; off > 0; off >>= 1) {
      if (t < off) red[t] = fmaxf(red[t], red[t + off]);
      __syncthreads();
    }
    float thr = red[0] * 0.3f;
    unsigned m = (s > thr) ? 1u : 0u;
    pref[t] = m;
    __syncthreads();
    for (int off = 1; off < 1024; off <<= 1) {
      unsigned v = (t >= off) ? pref[t - off] : 0u;
      __syncthreads();
      pref[t] += v;
      __syncthreads();
    }
    unsigned incl = pref[t];
    unsigned total = pref[1023];
    if (m) {
      int pos = count + (int)incl - 1;
      if (pos < 512) { i1o[b * 512 + pos] = n1; i2o[b * 512 + pos] = t; }
    }
    __syncthreads();
    if (t == 0) s_count = count + (int)total;
    __syncthreads();
  }
  __syncthreads();
  int cf = s_count;
  if (t < 512 && t >= cf) {  // pad: idx = N1*N2 -> i1 clamps to 1023, i2 = 0
    i1o[b * 512 + t] = 1023;
    i2o[b * 512 + t] = 0;
  }
}

// ---------------------------------------------------------------------------
// Kernel 2: gather corr = [feat1[i1] | feat2[i2]]  (f32 + f16 copies)
// grid = B*512, block = 128
// ---------------------------------------------------------------------------
__global__ void gather_kernel(const float* __restrict__ feat1, const float* __restrict__ feat2,
                              const int* __restrict__ i1, const int* __restrict__ i2,
                              float* __restrict__ corr, _Float16* __restrict__ corrh) {
  int row = blockIdx.x;
  int b = row >> 9;
  int t = threadIdx.x;
  int r1 = i1[row], r2 = i2[row];
  const float* s1 = feat1 + ((size_t)b * 1024 + r1) * 256;
  const float* s2 = feat2 + ((size_t)b * 1024 + r2) * 256;
  float* d = corr + (size_t)row * 512;
  _Float16* dh = corrh + (size_t)row * 512;
#pragma unroll
  for (int i = 0; i < 4; ++i) {
    int e = t + i * 128;
    float v = (e < 256) ? s1[e] : s2[e - 256];
    d[e] = v;
    dh[e] = (_Float16)v;
  }
}

// ---------------------------------------------------------------------------
// Kernel 3/5: Y = A(f16) @ W(f16,[N][K])^T + bias.  128x64 tile, 8 waves,
// double-buffered LDS.  grid = (M/128, N/64), block = 256
// ---------------------------------------------------------------------------
template <bool OUT_F16>
__global__ void gemm_xwT(const _Float16* __restrict__ A, const _Float16* __restrict__ W,
                         const float* __restrict__ bias, void* __restrict__ Y,
                         int M, int N, int K) {
  __shared__ _Float16 As[2][128 * 32];
  __shared__ _Float16 Bs[2][64 * 32];
  int m0 = blockIdx.x * 128;
  int n0 = blockIdx.y * 64;
  int t = threadIdx.x;         // 0..255
  int lane = t & 31;
  int wave = t >> 5;           // 0..7
  int wr = (wave >> 1) * 32;   // 0,32,64,96
  int wc = (wave & 1) * 32;    // 0,32

  int ar = t >> 1, aseg = (t & 1) * 16;           // A: 128 rows x 2 segs
  const _Float16* aptr = A + (size_t)(m0 + ar) * K + aseg;
  int br = t >> 1, bseg = (t & 1) * 16;           // B: 64 rows x 2 segs (t<128)
  const _Float16* bptr = W + (size_t)(n0 + br) * K + bseg;

  // preload tile 0
  *(uint4*)&As[0][ar * 32 + aseg] = *(const uint4*)(aptr);
  if (t < 128) *(uint4*)&Bs[0][br * 32 + bseg] = *(const uint4*)(bptr);
  __syncthreads();

  v8f c[2][2] = {};
  int nk = K / 32;
  for (int ki = 0; ki < nk; ++ki) {
    int buf = ki & 1;
    uint4 ra = {}, rb = {};
    bool more = (ki + 1 < nk);
    if (more) {                       // issue next tile's global loads (loadcnt)
      ra = *(const uint4*)(aptr + (ki + 1) * 32);
      if (t < 128) rb = *(const uint4*)(bptr + (ki + 1) * 32);
    }
    v16h a0 = load_a_frag(As[buf], wr + 0, 0, 32, lane);
    v16h a1 = load_a_frag(As[buf], wr + 16, 0, 32, lane);
    v16h b0 = load_b_frag_nk(Bs[buf], wc + 0, 0, 32, lane);
    v16h b1 = load_b_frag_nk(Bs[buf], wc + 16, 0, 32, lane);
    c[0][0] = wmma16x16x32(a0, b0, c[0][0]);
    c[0][1] = wmma16x16x32(a0, b1, c[0][1]);
    c[1][0] = wmma16x16x32(a1, b0, c[1][0]);
    c[1][1] = wmma16x16x32(a1, b1, c[1][1]);
    if (more) {                       // commit staged regs to other buffer
      *(uint4*)&As[buf ^ 1][ar * 32 + aseg] = ra;
      if (t < 128) *(uint4*)&Bs[buf ^ 1][br * 32 + bseg] = rb;
    }
    __syncthreads();
  }

#pragma unroll
  for (int i = 0; i < 2; ++i)
#pragma unroll
    for (int j = 0; j < 2; ++j) {
      int mbase = m0 + wr + i * 16 + ((lane & 16) ? 8 : 0);
      int ncol = n0 + wc + j * 16 + (lane & 15);
      float bv = bias[ncol];
#pragma unroll
      for (int v = 0; v < 8; ++v) {
        float val = c[i][j][v] + bv;
        size_t off = (size_t)(mbase + v) * N + ncol;
        if (OUT_F16) ((_Float16*)Y)[off] = (_Float16)val;
        else         ((float*)Y)[off] = val;
      }
    }
}

// ---------------------------------------------------------------------------
// Kernel 4: attention per (b, h, 32-q-row tile).  block = 64 (2 waves)
// qkv layout: [B*S][1536] f16, segments q|k|v of 512, head h at cols h*64..h*64+63
// ---------------------------------------------------------------------------
__global__ void attention_kernel(const _Float16* __restrict__ qkv,
                                 _Float16* __restrict__ attnh) {
  __shared__ _Float16 qs[32 * 64];
  __shared__ _Float16 ks[64 * 64];
  __shared__ _Float16 vs[32 * 64];
  __shared__ _Float16 sc[32 * 512];   // logits -> probs (f16)
  int blk = blockIdx.x;
  int qt = blk & 15;
  int h = (blk >> 4) & 7;
  int b = blk >> 7;
  int q0 = qt * 32;
  int t = threadIdx.x;
  int lane = t & 31;
  int wave = t >> 5;
  const size_t ldq = 1536;
  const _Float16* base = qkv + (size_t)b * 512 * ldq;

  {  // stage q tile 32x64
    int r = t >> 1, seg = (t & 1) * 32;
    const _Float16* src = base + (size_t)(q0 + r) * ldq + h * 64 + seg;
    *(uint4*)&qs[r * 64 + seg]      = *(const uint4*)(src);
    *(uint4*)&qs[r * 64 + seg + 16] = *(const uint4*)(src + 16);
  }
  __syncthreads();
  v16h qa0 = load_a_frag(qs, wave * 16, 0, 64, lane);
  v16h qa1 = load_a_frag(qs, wave * 16, 32, 64, lane);

  // phase 1: logits
  for (int kc = 0; kc < 8; ++kc) {
    {  // stage k chunk 64x64
      const _Float16* src = base + (size_t)(kc * 64 + t) * ldq + 512 + h * 64;
      *(uint4*)&ks[t * 64 + 0]  = *(const uint4*)(src + 0);
      *(uint4*)&ks[t * 64 + 16] = *(const uint4*)(src + 16);
      *(uint4*)&ks[t * 64 + 32] = *(const uint4*)(src + 32);
      *(uint4*)&ks[t * 64 + 48] = *(const uint4*)(src + 48);
    }
    __syncthreads();
#pragma unroll
    for (int j = 0; j < 4; ++j) {
      v8f acc = {};
      v16h kb0 = load_b_frag_nk(ks, j * 16, 0, 64, lane);
      v16h kb1 = load_b_frag_nk(ks, j * 16, 32, 64, lane);
      acc = wmma16x16x32(qa0, kb0, acc);
      acc = wmma16x16x32(qa1, kb1, acc);
      int row = wave * 16 + ((lane & 16) ? 8 : 0);
      int col = kc * 64 + j * 16 + (lane & 15);
#pragma unroll
      for (int v = 0; v < 8; ++v) sc[(size_t)(row + v) * 512 + col] = (_Float16)(acc[v] * 0.125f);
    }
    __syncthreads();
  }

  // phase 2: softmax — wave-parallel, each wave does its 16 rows, lanes split cols
  for (int r = 0; r < 16; ++r) {
    _Float16* row = &sc[(size_t)(wave * 16 + r) * 512];
    float v[16];
    float mx = -3.0e38f;
#pragma unroll
    for (int i = 0; i < 16; ++i) { v[i] = (float)row[i * 32 + lane]; mx = fmaxf(mx, v[i]); }
#pragma unroll
    for (int m = 16; m >= 1; m >>= 1) mx = fmaxf(mx, __shfl_xor(mx, m, 32));
    float sum = 0.f;
#pragma unroll
    for (int i = 0; i < 16; ++i) { v[i] = expf(v[i] - mx); sum += v[i]; }
#pragma unroll
    for (int m = 16; m >= 1; m >>= 1) sum += __shfl_xor(sum, m, 32);
    float inv = 1.f / sum;
#pragma unroll
    for (int i = 0; i < 16; ++i) row[i * 32 + lane] = (_Float16)(v[i] * inv);
  }
  __syncthreads();

  // phase 3: out = P @ V
  v8f o[4] = {};
  for (int vc = 0; vc < 16; ++vc) {
    {  // stage v chunk 32x64
      int r = t >> 1, seg = (t & 1) * 32;
      const _Float16* src = base + (size_t)(vc * 32 + r) * ldq + 1024 + h * 64 + seg;
      *(uint4*)&vs[r * 64 + seg]      = *(const uint4*)(src);
      *(uint4*)&vs[r * 64 + seg + 16] = *(const uint4*)(src + 16);
    }
    __syncthreads();
    v16h pa = load_a_frag(sc, wave * 16, vc * 32, 512, lane);
#pragma unroll
    for (int j = 0; j < 4; ++j) {
      v16h vb = load_b_frag_kn(vs, j * 16, 0, 64, lane);
      o[j] = wmma16x16x32(pa, vb, o[j]);
    }
    __syncthreads();
  }

#pragma unroll
  for (int j = 0; j < 4; ++j) {
    int col = h * 64 + j * 16 + (lane & 15);
    int rbase = q0 + wave * 16 + ((lane & 16) ? 8 : 0);
#pragma unroll
    for (int v = 0; v < 8; ++v)
      attnh[(size_t)(b * 512 + rbase + v) * 512 + col] = (_Float16)o[j][v];
  }
}

// ---------------------------------------------------------------------------
// Kernel 6: y = LN(corr + attn_out) (in-place into attn_out). grid=B*S, block=128
// ---------------------------------------------------------------------------
__global__ void ln_res_kernel(const float* __restrict__ corr, float* __restrict__ y,
                              const float* __restrict__ g, const float* __restrict__ bb) {
  __shared__ float red[128];
  int row = blockIdx.x, t = threadIdx.x;
  const float* c = corr + (size_t)row * 512;
  float* yr = y + (size_t)row * 512;
  float x[4], s = 0.f;
#pragma unroll
  for (int i = 0; i < 4; ++i) { x[i] = c[t + i * 128] + yr[t + i * 128]; s += x[i]; }
  float mean = block_reduce_sum(s, red, 128) * (1.f / 512.f);
  float vs = 0.f;
#pragma unroll
  for (int i = 0; i < 4; ++i) { float d = x[i] - mean; vs += d * d; }
  float var = block_reduce_sum(vs, red, 128) * (1.f / 512.f);
  float inv = rsqrtf(var + 1e-5f);
#pragma unroll
  for (int i = 0; i < 4; ++i) {
    int e = t + i * 128;
    yr[e] = (x[i] - mean) * inv * g[e] + bb[e];
  }
}

// Kernel 7: agg = mean over S. grid=B, block=512
__global__ void agg_kernel(const float* __restrict__ cl, float* __restrict__ agg) {
  int b = blockIdx.x, e = threadIdx.x;
  float s = 0.f;
  for (int ss = 0; ss < 512; ++ss) s += cl[((size_t)b * 512 + ss) * 512 + e];
  agg[b * 512 + e] = s * (1.f / 512.f);
}

// ---------------------------------------------------------------------------
// Kernel 8: MLP head + quaternion -> R.  grid=B, block=512
// out layout: R (B,3,3) flat [0..287], trans (B,3) flat [288..383]
// ---------------------------------------------------------------------------
__global__ void head_kernel(const float* __restrict__ agg,
                            const float* __restrict__ W1, const float* __restrict__ b1,
                            const float* __restrict__ g1, const float* __restrict__ bb1,
                            const float* __restrict__ W2, const float* __restrict__ b2,
                            const float* __restrict__ g2, const float* __restrict__ bb2,
                            const float* __restrict__ W3, const float* __restrict__ b3,
                            const float* __restrict__ W4, const float* __restrict__ b4,
                            float* __restrict__ out) {
  __shared__ float xa[512];
  __shared__ float h1[512];
  __shared__ float h2[256];
  __shared__ float h3[128];
  __shared__ float pose[10];
  __shared__ float red[512];
  int b = blockIdx.x, t = threadIdx.x;
  xa[t] = agg[b * 512 + t];
  __syncthreads();
  // layer 1: 512 -> 512, LN, gelu
  float acc = b1[t];
  for (int k = 0; k < 512; ++k) acc += xa[k] * W1[(size_t)t * 512 + k];
  float mean = block_reduce_sum(acc, red, 512) * (1.f / 512.f);
  float d = acc - mean;
  float var = block_reduce_sum(d * d, red, 512) * (1.f / 512.f);
  h1[t] = gelu_exact(d * rsqrtf(var + 1e-5f) * g1[t] + bb1[t]);
  __syncthreads();
  // layer 2: 512 -> 256, LN, gelu
  float acc2 = 0.f;
  if (t < 256) {
    acc2 = b2[t];
    for (int k = 0; k < 512; ++k) acc2 += h1[k] * W2[(size_t)t * 512 + k];
  }
  float m2 = block_reduce_sum((t < 256) ? acc2 : 0.f, red, 512) * (1.f / 256.f);
  float d2 = acc2 - m2;
  float v2 = block_reduce_sum((t < 256) ? d2 * d2 : 0.f, red, 512) * (1.f / 256.f);
  if (t < 256) h2[t] = gelu_exact(d2 * rsqrtf(v2 + 1e-5f) * g2[t] + bb2[t]);
  __syncthreads();
  // layer 3: 256 -> 128, gelu
  if (t < 128) {
    float a3 = b3[t];
    for (int k = 0; k < 256; ++k) a3 += h2[k] * W3[(size_t)t * 256 + k];
    h3[t] = gelu_exact(a3);
  }
  __syncthreads();
  // layer 4: 128 -> 10
  if (t < 10) {
    float a4 = b4[t];
    for (int k = 0; k < 128; ++k) a4 += h3[k] * W4[(size_t)t * 128 + k];
    pose[t] = a4;
  }
  __syncthreads();
  if (t == 0) {
    float w = pose[0], x = pose[1], y = pose[2], z = pose[3];
    float qn = fmaxf(sqrtf(w * w + x * x + y * y + z * z), 1e-12f);
    w /= qn; x /= qn; y /= qn; z /= qn;
    float tx = pose[4], ty = pose[5], tz = pose[6];
    float tn = fmaxf(sqrtf(tx * tx + ty * ty + tz * tz), 1e-12f);
    float* R = out + b * 9;
    R[0] = 1.f - 2.f * (y * y + z * z); R[1] = 2.f * (x * y - w * z); R[2] = 2.f * (x * z + w * y);
    R[3] = 2.f * (x * y + w * z); R[4] = 1.f - 2.f * (x * x + z * z); R[5] = 2.f * (y * z - w * x);
    R[6] = 2.f * (x * z - w * y); R[7] = 2.f * (y * z + w * x); R[8] = 1.f - 2.f * (x * x + y * y);
    float* T = out + 288 + b * 3;
    T[0] = tx / tn; T[1] = ty / tn; T[2] = tz / tn;
  }
}

// ---------------------------------------------------------------------------
extern "C" void kernel_launch(void* const* d_in, const int* in_sizes, int n_in,
                              void* d_out, int out_size, void* d_ws, size_t ws_size,
                              hipStream_t stream) {
  (void)in_sizes; (void)n_in; (void)out_size; (void)ws_size;
  const float* feat1 = (const float*)d_in[2];
  const float* feat2 = (const float*)d_in[3];
  const float* match = (const float*)d_in[4];
  const float* w_qkv = (const float*)d_in[6];
  const float* b_qkv = (const float*)d_in[7];
  const float* w_o   = (const float*)d_in[8];
  const float* b_o   = (const float*)d_in[9];
  const float* ln0_g = (const float*)d_in[10];
  const float* ln0_b = (const float*)d_in[11];
  const float* W1 = (const float*)d_in[12];
  const float* b1 = (const float*)d_in[13];
  const float* g1 = (const float*)d_in[14];
  const float* bb1 = (const float*)d_in[15];
  const float* W2 = (const float*)d_in[16];
  const float* b2 = (const float*)d_in[17];
  const float* g2 = (const float*)d_in[18];
  const float* bb2 = (const float*)d_in[19];
  const float* W3 = (const float*)d_in[20];
  const float* b3 = (const float*)d_in[21];
  const float* W4 = (const float*)d_in[22];
  const float* b4 = (const float*)d_in[23];
  float* out = (float*)d_out;

  char* p = (char*)d_ws;
  auto alloc = [&](size_t bytes) {
    char* r = p;
    p += (bytes + 255) & ~(size_t)255;
    return r;
  };
  int*      i1    = (int*)alloc((size_t)32 * 512 * 4);
  int*      i2    = (int*)alloc((size_t)32 * 512 * 4);
  float*    corr  = (float*)alloc((size_t)32 * 512 * 512 * 4);
  _Float16* corrh = (_Float16*)alloc((size_t)32 * 512 * 512 * 2);
  _Float16* qkvh  = (_Float16*)alloc((size_t)32 * 512 * 1536 * 2);
  _Float16* attnh = (_Float16*)alloc((size_t)32 * 512 * 512 * 2);
  float*    attno = (float*)alloc((size_t)32 * 512 * 512 * 4);
  float*    aggb  = (float*)alloc((size_t)32 * 512 * 4);
  _Float16* wqkvh = (_Float16*)alloc((size_t)1536 * 512 * 2);
  _Float16* woh   = (_Float16*)alloc((size_t)512 * 512 * 2);

  const int nwqkv = 1536 * 512, nwo = 512 * 512;
  cvt_f16_kernel<<<(nwqkv / 4 + 255) / 256, 256, 0, stream>>>(w_qkv, wqkvh, nwqkv);
  cvt_f16_kernel<<<(nwo / 4 + 255) / 256, 256, 0, stream>>>(w_o, woh, nwo);

  topk_kernel<<<32, 1024, 0, stream>>>(match, i1, i2);
  gather_kernel<<<32 * 512, 128, 0, stream>>>(feat1, feat2, i1, i2, corr, corrh);
  gemm_xwT<true><<<dim3(16384 / 128, 1536 / 64), 256, 0, stream>>>(
      corrh, wqkvh, b_qkv, (void*)qkvh, 16384, 1536, 512);
  attention_kernel<<<32 * 8 * 16, 64, 0, stream>>>(qkvh, attnh);
  gemm_xwT<false><<<dim3(16384 / 128, 512 / 64), 256, 0, stream>>>(
      attnh, woh, b_o, (void*)attno, 16384, 512, 512);
  ln_res_kernel<<<16384, 128, 0, stream>>>(corr, attno, ln0_g, ln0_b);
  agg_kernel<<<32, 512, 0, stream>>>(attno, aggb);
  head_kernel<<<32, 512, 0, stream>>>(aggb, W1, b1, g1, bb1, W2, b2, g2, bb2,
                                      W3, b3, W4, b4, out);
}
